// Head_10204842295965
// MI455X (gfx1250) — matrix operand order repaired and checked by reading, not compiled, not run on hardware
//
#include <hip/hip_runtime.h>
#include <hip/hip_bf16.h>

typedef _Float16 f16;
typedef __attribute__((ext_vector_type(16))) _Float16 v16h;
typedef __attribute__((ext_vector_type(8)))  _Float16 v8h;
typedef __attribute__((ext_vector_type(8)))  float    v8f;
typedef __attribute__((ext_vector_type(4)))  unsigned int u32x4;
typedef __attribute__((ext_vector_type(8)))  int          i32x8;
typedef __attribute__((ext_vector_type(4)))  int          i32x4;

#define BB 8
#define TT 4096
#define CC 1024
#define HH 128
#define KT 64   // keys per tile in attention kernel

#if __has_builtin(__builtin_amdgcn_tensor_load_to_lds)
#define HAVE_TDM 1
#else
#define HAVE_TDM 0
#endif

static __device__ __forceinline__ v8f wmma16(v16h a, v16h b, v8f c) {
  // D = A(16x32 f16) * B(32x16 f16) + C(16x16 f32)
  return __builtin_amdgcn_wmma_f32_16x16x32_f16(false, a, false, b, (short)0, c,
                                                false, false);
}

static __device__ __forceinline__ void wait_tensorcnt0() {
#if __has_builtin(__builtin_amdgcn_s_wait_tensorcnt)
  __builtin_amdgcn_s_wait_tensorcnt(0);
#else
  asm volatile("s_wait_tensorcnt 0x0" ::: "memory");
#endif
}

// Issue a 1-D contiguous TDM copy (nbytes, multiple of 8) global -> LDS.
// Must be executed by exactly one wave; caller waits tensorcnt + barrier.
static __device__ __forceinline__ void tdm_copy_1d(unsigned lds_addr,
                                                   const void* gaddr,
                                                   unsigned n8 /* 8B units */) {
#if HAVE_TDM
  unsigned long long ga = (unsigned long long)(size_t)gaddr;
  u32x4 g0;
  g0[0] = 1u;                                   // count=1, user mode
  g0[1] = lds_addr;                             // lds_addr (bytes)
  g0[2] = (unsigned)(ga & 0xFFFFFFFFu);         // global_addr[31:0]
  g0[3] = (unsigned)((ga >> 32) & 0x01FFFFFFu)  // global_addr[56:32]
          | (2u << 30);                         // type = 2 ("image")
  i32x8 g1 = {};
  g1[0] = (int)(3u << 16);                      // data_size = 3 (8 bytes)
  g1[1] = (int)(n8 << 16);                      // tensor_dim0[15:0] @ bits63:48
  g1[2] = (int)(1u << 16);                      // tensor_dim1 = 1
  g1[3] = (int)(n8 << 16);                      // tile_dim0 @ bits127:112
  g1[4] = 0;                                    // tile_dim1/2 = 0 (unused)
  g1[5] = (int)n8;                              // tensor_dim0_stride (8B units)
  g1[6] = 0;
  g1[7] = 0;
  i32x4 gz = {};
#if defined(__clang_major__) && (__clang_major__ >= 23)
  i32x8 gz8 = {};
  __builtin_amdgcn_tensor_load_to_lds(g0, g1, gz, gz, gz8, 0);
#else
  __builtin_amdgcn_tensor_load_to_lds(g0, g1, gz, gz, 0);
#endif
#else
  (void)lds_addr; (void)gaddr; (void)n8;
#endif
}

// ---------------------------------------------------------------------------
// Kernel 1: QKV projection.  out[row, h] = sum_c x[row, c] * W[h, c]
// One block = 128 rows x 128 heads; blockIdx.y selects {Q, K, V}.
// W K-slab staged transposed into LDS as f16, shared by all 8 waves.
// ---------------------------------------------------------------------------
__global__ __launch_bounds__(256) void head_proj_qkv(
    const float* __restrict__ x,
    const float* __restrict__ Wq, const float* __restrict__ Wk,
    const float* __restrict__ Wv,
    f16* __restrict__ qg, f16* __restrict__ kg, f16* __restrict__ vg)
{
  __shared__ alignas(64) f16 Wt[HH * 32];   // transposed slab: Wt[n*32 + k]

  const float* W = (blockIdx.y == 0) ? Wq : (blockIdx.y == 1) ? Wk : Wv;
  f16* out       = (blockIdx.y == 0) ? qg : (blockIdx.y == 1) ? kg : vg;

  const int tid  = threadIdx.x;
  const int wv   = tid >> 5;      // wave id 0..7
  const int lane = tid & 31;
  const int hw   = lane >> 4;     // half-wave (0/1)
  const int l16  = lane & 15;
  const int row0 = blockIdx.x * 128 + wv * 16;
  const int rowA = row0 + l16;    // A-matrix row for this lane

  v8f acc[8] = {};

  for (int kb = 0; kb < CC; kb += 32) {
    __syncthreads();
    {
      // 256 threads stage W[0..127][kb..kb+31] -> Wt (transposed, f16)
      const int n  = tid >> 1;
      const int kh = (tid & 1) * 16;
      const float4* src = (const float4*)(W + (size_t)n * CC + kb + kh);
      float4 w0 = src[0], w1 = src[1], w2 = src[2], w3 = src[3];
      f16* dst = &Wt[n * 32 + kh];
      dst[ 0] = (f16)w0.x; dst[ 1] = (f16)w0.y; dst[ 2] = (f16)w0.z; dst[ 3] = (f16)w0.w;
      dst[ 4] = (f16)w1.x; dst[ 5] = (f16)w1.y; dst[ 6] = (f16)w1.z; dst[ 7] = (f16)w1.w;
      dst[ 8] = (f16)w2.x; dst[ 9] = (f16)w2.y; dst[10] = (f16)w2.z; dst[11] = (f16)w2.w;
      dst[12] = (f16)w3.x; dst[13] = (f16)w3.y; dst[14] = (f16)w3.z; dst[15] = (f16)w3.w;
    }
    __syncthreads();

    // A fragment (16x32 f16): lane row = l16; half 0 -> K {0..15},
    // half 1 -> K {16..31}  (contiguous order, matches B)
    const float* xr = x + (size_t)rowA * CC + kb;
    const float4* xa = (const float4*)(xr + hw * 16);
    float4 a0 = xa[0], a1 = xa[1];
    const float4* xb = (const float4*)(xr + 8 + hw * 16);
    float4 a2 = xb[0], a3 = xb[1];
    v16h a;
    a[ 0] = (f16)a0.x; a[ 1] = (f16)a0.y; a[ 2] = (f16)a0.z; a[ 3] = (f16)a0.w;
    a[ 4] = (f16)a1.x; a[ 5] = (f16)a1.y; a[ 6] = (f16)a1.z; a[ 7] = (f16)a1.w;
    a[ 8] = (f16)a2.x; a[ 9] = (f16)a2.y; a[10] = (f16)a2.z; a[11] = (f16)a2.w;
    a[12] = (f16)a3.x; a[13] = (f16)a3.y; a[14] = (f16)a3.z; a[15] = (f16)a3.w;

    // Preload all 8 B fragments, then stream 8 WMMAs back-to-back (XDL pipe).
    v16h bf[8];
#pragma unroll
    for (int nt = 0; nt < 8; ++nt)
      bf[nt] = *(const v16h*)&Wt[(nt * 16 + l16) * 32 + hw * 16];
#pragma unroll
    for (int nt = 0; nt < 8; ++nt)
      acc[nt] = wmma16(a, bf[nt], acc[nt]);
  }

  // C/D layout: VGPR r, lane -> (M = r + 8*hw, N = l16)
#pragma unroll
  for (int nt = 0; nt < 8; ++nt)
#pragma unroll
    for (int r = 0; r < 8; ++r) {
      int row = row0 + hw * 8 + r;
      out[(size_t)row * HH + nt * 16 + l16] = (f16)acc[nt][r];
    }
}

// ---------------------------------------------------------------------------
// Kernel 2: causal flash attention.  Each block: 128 query rows of one batch.
// Each wave: 16 query rows.  K tile (contiguous 16KB) streamed to LDS by the
// Tensor Data Mover (wave 0) while all waves stage V transposed; online
// softmax; P -> LDS round-trip for the A-layout of the second GEMM.
// ---------------------------------------------------------------------------
__global__ __launch_bounds__(256) void head_flash_attn(
    const f16* __restrict__ qg, const f16* __restrict__ kg,
    const f16* __restrict__ vg, float* __restrict__ out)
{
  __shared__ alignas(64) f16 Ks[KT * HH];       // [key][h]   16 KB
  __shared__ alignas(64) f16 Vs[HH * KT];       // [h][key]   16 KB (transposed)
  __shared__ alignas(64) f16 Ps[8 * 16 * KT];   // per-wave P [16][64] 16 KB

  const int tid  = threadIdx.x;
  const int wv   = tid >> 5;
  const int lane = tid & 31;
  const int hw   = lane >> 4;
  const int l16  = lane & 15;
  const int b    = blockIdx.y;
  const int qi   = blockIdx.x;                  // query tile (128 rows)

  const size_t baseT = (size_t)b * TT;
  const float  scale = 0.08838834764831845f;    // 1/sqrt(128)
  const unsigned ldsKs = (unsigned)(size_t)(const void*)&Ks[0]; // LDS byte addr

  // Q fragments: 4 K-steps of 32 over H=128, pre-scaled
  v16h qf[4];
  {
    const f16* qrow = qg + (baseT + (size_t)(qi * 128 + wv * 16 + l16)) * HH;
#pragma unroll
    for (int ks = 0; ks < 4; ++ks) {
      v8h h1 = *(const v8h*)(qrow + ks * 32 + hw * 16);
      v8h h2 = *(const v8h*)(qrow + ks * 32 + 8 + hw * 16);
#pragma unroll
      for (int i = 0; i < 8; ++i) {
        qf[ks][i]     = (f16)((float)h1[i] * scale);
        qf[ks][8 + i] = (f16)((float)h2[i] * scale);
      }
    }
  }

  v8f   o[8] = {};
  float m_run[8], l_run[8];
#pragma unroll
  for (int r = 0; r < 8; ++r) { m_run[r] = -3.0e38f; l_run[r] = 0.f; }

  const int q_t0   = qi * 128 + wv * 16 + hw * 8;  // + r = absolute query pos
  const int jt_end = 2 * qi + 2;                   // causal: keys <= qi*128+127

  for (int jt = 0; jt < jt_end; ++jt) {
    __syncthreads();   // protect Ks/Vs/Ps reads of previous iteration

    const size_t gtile = (baseT + (size_t)(jt * KT)) * HH;   // tile start elem

#if HAVE_TDM
    if (wv == 0) {
      // K tile is a contiguous 16KB block: 1-D TDM transfer, async.
      tdm_copy_1d(ldsKs, (const void*)(kg + gtile), (KT * HH * 2) / 8);
    }
#endif
    {
      // all 256 threads stage V transposed (and K manually if no TDM)
      const int key = tid >> 2;
      const int hh  = (tid & 3) * 32;
      const size_t g = gtile + (size_t)key * HH + hh;
#if !HAVE_TDM
      const v8h* kp = (const v8h*)(kg + g);
      v8h k0 = kp[0], k1 = kp[1], k2 = kp[2], k3 = kp[3];
      *(v8h*)&Ks[key * HH + hh +  0] = k0;
      *(v8h*)&Ks[key * HH + hh +  8] = k1;
      *(v8h*)&Ks[key * HH + hh + 16] = k2;
      *(v8h*)&Ks[key * HH + hh + 24] = k3;
#endif
      const v8h* vp = (const v8h*)(vg + g);
      v8h v0 = vp[0], v1 = vp[1], v2 = vp[2], v3 = vp[3];
#pragma unroll
      for (int j = 0; j < 8; ++j) {
        Vs[(hh +  0 + j) * KT + key] = v0[j];
        Vs[(hh +  8 + j) * KT + key] = v1[j];
        Vs[(hh + 16 + j) * KT + key] = v2[j];
        Vs[(hh + 24 + j) * KT + key] = v3[j];
      }
    }
#if HAVE_TDM
    if (wv == 0) wait_tensorcnt0();
#endif
    __syncthreads();

    // S = Q * K^T   (16 x 64 per wave, 4 tiles of 16x16)
    v8f s[4];
#pragma unroll
    for (int nt = 0; nt < 4; ++nt) {
      v16h kf[4];
#pragma unroll
      for (int ks = 0; ks < 4; ++ks)
        kf[ks] = *(const v16h*)&Ks[(nt * 16 + l16) * HH + ks * 32 + hw * 16];
      v8f accS = {};
#pragma unroll
      for (int ks = 0; ks < 4; ++ks)
        accS = wmma16(qf[ks], kf[ks], accS);
      s[nt] = accS;
    }

    // causal mask + per-row running max
    float rm[8];
#pragma unroll
    for (int r = 0; r < 8; ++r) rm[r] = -3.0e38f;
#pragma unroll
    for (int nt = 0; nt < 4; ++nt) {
      const int key_t = jt * KT + nt * 16 + l16;
#pragma unroll
      for (int r = 0; r < 8; ++r) {
        float vS = s[nt][r];
        if (key_t > q_t0 + r) vS = -3.0e38f;
        s[nt][r] = vS;
        rm[r] = fmaxf(rm[r], vS);
      }
    }
#pragma unroll
    for (int r = 0; r < 8; ++r)
      for (int m = 1; m < 16; m <<= 1)
        rm[r] = fmaxf(rm[r], __shfl_xor(rm[r], m, 32));

    // online softmax update
    float alpha[8], rs[8];
#pragma unroll
    for (int r = 0; r < 8; ++r) {
      float nm  = fmaxf(m_run[r], rm[r]);
      alpha[r]  = __expf(m_run[r] - nm);
      m_run[r]  = nm;
      rs[r]     = 0.f;
    }
#pragma unroll
    for (int nt = 0; nt < 4; ++nt)
#pragma unroll
      for (int r = 0; r < 8; ++r) {
        float p = __expf(s[nt][r] - m_run[r]);
        rs[r] += p;
        Ps[wv * (16 * KT) + (hw * 8 + r) * KT + nt * 16 + l16] = (f16)p;
      }
#pragma unroll
    for (int r = 0; r < 8; ++r) {
      for (int m = 1; m < 16; m <<= 1) rs[r] += __shfl_xor(rs[r], m, 32);
      l_run[r] = l_run[r] * alpha[r] + rs[r];
    }
#pragma unroll
    for (int nt = 0; nt < 8; ++nt)
#pragma unroll
      for (int r = 0; r < 8; ++r) o[nt][r] *= alpha[r];

    __syncthreads();   // make P stores visible / ordered before A-layout reads

    // O += P * V   (P: 16x64 A-layout via LDS; V transposed in LDS)
#pragma unroll
    for (int ks = 0; ks < 2; ++ks) {
      const f16* prow = &Ps[wv * (16 * KT) + l16 * KT + ks * 32 + hw * 16];
      v8h c1 = *(const v8h*)prow;
      v8h c2 = *(const v8h*)(prow + 8);
      v16h pa;
#pragma unroll
      for (int i = 0; i < 8; ++i) { pa[i] = c1[i]; pa[8 + i] = c2[i]; }
      v16h vf[8];
#pragma unroll
      for (int nt = 0; nt < 8; ++nt)
        vf[nt] = *(const v16h*)&Vs[(nt * 16 + l16) * KT + ks * 32 + hw * 16];
#pragma unroll
      for (int nt = 0; nt < 8; ++nt)
        o[nt] = wmma16(pa, vf[nt], o[nt]);
    }
  }

  // normalize and store fp32 output [B,T,H]
  float inv[8];
#pragma unroll
  for (int r = 0; r < 8; ++r) inv[r] = 1.0f / l_run[r];
#pragma unroll
  for (int nt = 0; nt < 8; ++nt)
#pragma unroll
    for (int r = 0; r < 8; ++r) {
      size_t row = baseT + (size_t)(qi * 128 + wv * 16 + hw * 8 + r);
      out[row * HH + nt * 16 + l16] = o[nt][r] * inv[r];
    }
}

// ---------------------------------------------------------------------------
extern "C" void kernel_launch(void* const* d_in, const int* in_sizes, int n_in,
                              void* d_out, int out_size, void* d_ws, size_t ws_size,
                              hipStream_t stream) {
  (void)in_sizes; (void)n_in; (void)out_size; (void)ws_size;
  const float* x  = (const float*)d_in[0];
  const float* Wk = (const float*)d_in[1];
  const float* Wq = (const float*)d_in[2];
  const float* Wv = (const float*)d_in[3];

  const size_t qkv_bytes = (size_t)BB * TT * HH * sizeof(f16);  // 8 MB each
  f16* qg = (f16*)d_ws;
  f16* kg = (f16*)((char*)d_ws + qkv_bytes);
  f16* vg = (f16*)((char*)d_ws + 2 * qkv_bytes);

  dim3 gProj((BB * TT) / 128, 3);
  head_proj_qkv<<<gProj, 256, 0, stream>>>(x, Wq, Wk, Wv, qg, kg, vg);

  dim3 gAttn(TT / 128, BB);
  head_flash_attn<<<gAttn, 256, 0, stream>>>(qg, kg, vg, (float*)d_out);
}
